// Graph_inter_9509057593846
// MI455X (gfx1250) — compile-verified, matrix operand order checked by reading, not run on hardware
//
#include <hip/hip_runtime.h>
#include <hip/hip_bf16.h>

typedef __attribute__((ext_vector_type(16))) _Float16 v16h;
typedef __attribute__((ext_vector_type(8)))  _Float16 v8h;
typedef __attribute__((ext_vector_type(8)))  float    v8f;

#define DIMF 128     // feature dim (in and out, H*C)
#define HEADS 8
#define CPH 16
#define NEG_SLOPE 0.2f

// ---------------------------------------------------------------------------
// order-preserving float <-> uint encoding for atomicMax on floats
// ---------------------------------------------------------------------------
__device__ __forceinline__ unsigned f2ord(float f) {
    unsigned u = __float_as_uint(f);
    return (u & 0x80000000u) ? ~u : (u | 0x80000000u);
}
__device__ __forceinline__ float ord2f(unsigned u) {
    return (u & 0x80000000u) ? __uint_as_float(u & 0x7fffffffu)
                             : __uint_as_float(~u);
}
// ord encoding of -inf: bits(–inf)=0xFF800000 -> ~ = 0x007FFFFF
#define ORD_NEG_INF 0x007FFFFFu

// ---------------------------------------------------------------------------
// GEMM: Y[N x 128] = relu(X[N x 128]) @ W[128 x 128], f16 WMMA, f32 accum.
// Block = 256 threads = 8 waves; each wave owns 16 rows; block covers 128 rows.
// W is staged into LDS transposed ([n][k], f16) so B fragments are two
// contiguous 16B LDS loads per lane.
// ---------------------------------------------------------------------------
__global__ __launch_bounds__(256) void gemm_relu_wmma(
        const float* __restrict__ X, const float* __restrict__ W,
        float* __restrict__ Y, int nrows)
{
    __shared__ alignas(16) _Float16 WT[128 * 136];   // [n][k], padded stride

    const int tid = threadIdx.x;
    // cooperative transpose-convert of W (k-major 128x128) into WT[n][k]
    for (int idx = tid; idx < 128 * 128; idx += 256) {
        int k = idx >> 7, n = idx & 127;
        WT[n * 136 + k] = (_Float16)W[idx];
    }
    __syncthreads();

    const int wave = tid >> 5;
    const int lane = tid & 31;
    const int half = lane >> 4;     // which 16-lane half
    const int lm   = lane & 15;
    const int row0 = blockIdx.x * 128 + wave * 16;

    v8f acc[8];
#pragma unroll
    for (int t = 0; t < 8; ++t) acc[t] = v8f{0.f,0.f,0.f,0.f,0.f,0.f,0.f,0.f};

    int m = row0 + lm;
    int mclamp = m < nrows ? m : 0;
    const float* xrow = X + (size_t)mclamp * DIMF;

#pragma unroll
    for (int kk = 0; kk < 4; ++kk) {                 // K = 128 in steps of 32
        const int k0 = kk * 32 + half * 8;
        // A fragment: elems 0..7 -> k0..k0+7 ; elems 8..15 -> k0+16..k0+23
        float4 f0 = *(const float4*)(xrow + k0);
        float4 f1 = *(const float4*)(xrow + k0 + 4);
        float4 f2 = *(const float4*)(xrow + k0 + 16);
        float4 f3 = *(const float4*)(xrow + k0 + 20);
        float av[16] = { f0.x, f0.y, f0.z, f0.w, f1.x, f1.y, f1.z, f1.w,
                         f2.x, f2.y, f2.z, f2.w, f3.x, f3.y, f3.z, f3.w };
        v16h a;
#pragma unroll
        for (int i = 0; i < 16; ++i) {
            float v = av[i];
            a[i] = (_Float16)(v > 0.f ? v : 0.f);    // fused relu
        }
#pragma unroll
        for (int t = 0; t < 8; ++t) {                // 8 x 16-wide N tiles
            const int n = t * 16 + lm;
            const _Float16* wp = &WT[n * 136 + k0];
            v8h lo = *(const v8h*)(wp);
            v8h hi = *(const v8h*)(wp + 16);
            v16h b;
#pragma unroll
            for (int i = 0; i < 8; ++i) { b[i] = lo[i]; b[i + 8] = hi[i]; }
            acc[t] = __builtin_amdgcn_wmma_f32_16x16x32_f16(
                false, a, false, b, (short)0, acc[t], false, false);
        }
    }

    // C/D layout: VGPR j -> row = j (lanes 0-15) / j+8 (lanes 16-31), col = lm
#pragma unroll
    for (int t = 0; t < 8; ++t) {
        const int n = t * 16 + lm;
#pragma unroll
        for (int j = 0; j < 8; ++j) {
            int mm = row0 + half * 8 + j;
            if (mm < nrows) Y[(size_t)mm * DIMF + n] = acc[t][j];
        }
    }
}

// ---------------------------------------------------------------------------
// attention scores: a[n,h] = sum_c x[n, h*16+c] * att[h,c]
// ---------------------------------------------------------------------------
__global__ void att_scores(const float* __restrict__ Xf,
                           const float* __restrict__ att,
                           float* __restrict__ out, int n)
{
    int idx = blockIdx.x * blockDim.x + threadIdx.x;
    if (idx >= n * HEADS) return;
    int node = idx >> 3, h = idx & 7;
    const float4* xp = (const float4*)(Xf + (size_t)node * DIMF + h * CPH);
    const float4* ap = (const float4*)(att + h * CPH);
    float s = 0.f;
#pragma unroll
    for (int q = 0; q < 4; ++q) {
        float4 x = xp[q], a = ap[q];
        s += x.x * a.x + x.y * a.y + x.z * a.z + x.w * a.w;
    }
    out[idx] = s;
}

// ---------------------------------------------------------------------------
// init: out = bias broadcast, max = ord(-inf), den = 0
// ---------------------------------------------------------------------------
__global__ void init_dir(float* __restrict__ out, unsigned* __restrict__ mord,
                         float* __restrict__ den, const float* __restrict__ bias,
                         int n)
{
    int idx = blockIdx.x * blockDim.x + threadIdx.x;
    if (idx < n * DIMF) out[idx] = bias[idx & (DIMF - 1)];
    if (idx < n * HEADS) { mord[idx] = ORD_NEG_INF; den[idx] = 0.f; }
}

// ---------------------------------------------------------------------------
// edge helpers: one thread per (edge, head); edges E.. E+N-1 are self loops
// ---------------------------------------------------------------------------
__device__ __forceinline__ bool edge_sd(int t, const int* __restrict__ sArr,
                                        const int* __restrict__ dArr,
                                        int E, int N, int& e, int& h,
                                        int& s, int& d)
{
    if (t >= (E + N) * HEADS) return false;
    e = t >> 3; h = t & 7;
    if (e < E) { s = sArr[e]; d = dArr[e]; }
    else       { s = d = e - E; }
    return true;
}

__device__ __forceinline__ float edge_logit(const float* a_s, const float* a_d,
                                            int s, int d, int h)
{
    float v = a_s[s * HEADS + h] + a_d[d * HEADS + h];
    return v > 0.f ? v : NEG_SLOPE * v;
}

__global__ void edge_max(const int* __restrict__ sArr, const int* __restrict__ dArr,
                         const float* __restrict__ a_s, const float* __restrict__ a_d,
                         unsigned* __restrict__ mord, int E, int N)
{
    int t = blockIdx.x * blockDim.x + threadIdx.x;
    int e, h, s, d;
    if (!edge_sd(t, sArr, dArr, E, N, e, h, s, d)) return;
    atomicMax(&mord[d * HEADS + h], f2ord(edge_logit(a_s, a_d, s, d, h)));
}

__global__ void edge_den(const int* __restrict__ sArr, const int* __restrict__ dArr,
                         const float* __restrict__ a_s, const float* __restrict__ a_d,
                         const unsigned* __restrict__ mord, float* __restrict__ den,
                         int E, int N)
{
    int t = blockIdx.x * blockDim.x + threadIdx.x;
    int e, h, s, d;
    if (!edge_sd(t, sArr, dArr, E, N, e, h, s, d)) return;
    float v  = edge_logit(a_s, a_d, s, d, h);
    float mv = ord2f(mord[d * HEADS + h]);
    atomicAdd(&den[d * HEADS + h], __expf(v - mv));
}

__global__ void edge_agg(const int* __restrict__ sArr, const int* __restrict__ dArr,
                         const float* __restrict__ a_s, const float* __restrict__ a_d,
                         const unsigned* __restrict__ mord, const float* __restrict__ den,
                         const float* __restrict__ xs, float* __restrict__ out,
                         int E, int N)
{
    int t = blockIdx.x * blockDim.x + threadIdx.x;
    int e, h, s, d;
    if (!edge_sd(t, sArr, dArr, E, N, e, h, s, d)) return;
    float v     = edge_logit(a_s, a_d, s, d, h);
    float mv    = ord2f(mord[d * HEADS + h]);
    float alpha = __expf(v - mv) / den[d * HEADS + h];
    const float4* xp = (const float4*)(xs + (size_t)s * DIMF + h * CPH);
    float* op = out + (size_t)d * DIMF + h * CPH;
#pragma unroll
    for (int q = 0; q < 4; ++q) {
        float4 x = xp[q];
        atomicAdd(op + q * 4 + 0, x.x * alpha);
        atomicAdd(op + q * 4 + 1, x.y * alpha);
        atomicAdd(op + q * 4 + 2, x.z * alpha);
        atomicAdd(op + q * 4 + 3, x.w * alpha);
    }
}

// ---------------------------------------------------------------------------
extern "C" void kernel_launch(void* const* d_in, const int* in_sizes, int n_in,
                              void* d_out, int out_size, void* d_ws, size_t ws_size,
                              hipStream_t stream)
{
    const float* x_h     = (const float*)d_in[0];
    const float* x_t     = (const float*)d_in[1];
    const int*   ei      = (const int*)  d_in[2];   // [2, E] (src row, dst row)
    const float* W_src   = (const float*)d_in[3];
    const float* W_dst   = (const float*)d_in[4];
    const float* att_src = (const float*)d_in[5];
    const float* att_dst = (const float*)d_in[6];
    const float* bias    = (const float*)d_in[7];

    const int N = in_sizes[0] / DIMF;   // 50000
    const int E = in_sizes[2] / 2;      // 800000

    // workspace layout (reused across the two directions): ~58 MB
    float*    xs   = (float*)d_ws;                    // N*128
    float*    xd   = xs  + (size_t)N * DIMF;          // N*128
    float*    a_s  = xd  + (size_t)N * DIMF;          // N*8
    float*    a_d  = a_s + (size_t)N * HEADS;         // N*8
    unsigned* mord = (unsigned*)(a_d + (size_t)N * HEADS); // N*8
    float*    den  = (float*)(mord + (size_t)N * HEADS);   // N*8

    float* h_rep = (float*)d_out;                    // outputs: (h_rep, t_rep)
    float* t_rep = (float*)d_out + (size_t)N * DIMF;

    const int gemm_blocks = (N + 127) / 128;
    const int na = N * HEADS;
    const int ne = (E + N) * HEADS;

    for (int dir = 0; dir < 2; ++dir) {
        // dir 0: t_rep = gat(x_src=relu(x_h), x_dst=relu(x_t), src, dst)
        // dir 1: h_rep = gat(x_src=relu(x_t), x_dst=relu(x_h), dst, src)
        const float* xsrc = (dir == 0) ? x_h : x_t;
        const float* xdst = (dir == 0) ? x_t : x_h;
        const int*   sArr = (dir == 0) ? ei     : ei + E;
        const int*   dArr = (dir == 0) ? ei + E : ei;
        float*       outp = (dir == 0) ? t_rep  : h_rep;

        gemm_relu_wmma<<<gemm_blocks, 256, 0, stream>>>(xsrc, W_src, xs, N);
        gemm_relu_wmma<<<gemm_blocks, 256, 0, stream>>>(xdst, W_dst, xd, N);

        att_scores<<<(na + 255) / 256, 256, 0, stream>>>(xs, att_src, a_s, N);
        att_scores<<<(na + 255) / 256, 256, 0, stream>>>(xd, att_dst, a_d, N);

        init_dir<<<(N * DIMF + 255) / 256, 256, 0, stream>>>(outp, mord, den, bias, N);

        edge_max<<<(ne + 255) / 256, 256, 0, stream>>>(sArr, dArr, a_s, a_d, mord, E, N);
        edge_den<<<(ne + 255) / 256, 256, 0, stream>>>(sArr, dArr, a_s, a_d, mord, den, E, N);
        edge_agg<<<(ne + 255) / 256, 256, 0, stream>>>(sArr, dArr, a_s, a_d, mord, den,
                                                       xs, outp, E, N);
    }
}
